// AlpamayoVisionEncoder_90297392431320
// MI455X (gfx1250) — matrix operand order repaired and checked by reading, not compiled, not run on hardware
//
#include <hip/hip_runtime.h>
#include <hip/hip_bf16.h>
#include <cstdint>
#include <cmath>

typedef __bf16 bf16;
typedef __attribute__((ext_vector_type(16))) __bf16 v16bf;
typedef __attribute__((ext_vector_type(8)))  float  v8f;
typedef __attribute__((ext_vector_type(4))) unsigned int v4u;
typedef __attribute__((ext_vector_type(8)))  int v8i;
typedef __attribute__((ext_vector_type(4)))  int v4i;

#define NBATCH 4
#define SEQ    784
#define SEQP   800          // SEQ padded to mult of 32
#define HID_   1152
#define H3_    3456
#define FI_    4304
#define FIP_   4320         // FI padded to mult of 32
#define H4_    4608
#define OUTD_  4096
#define MERG   196
#define MTOT   3136         // NBATCH*SEQ (mult of 32)
#define MME    784          // NBATCH*MERG merger rows
#define MMEP   800
#define PK_    768          // patch GEMM K = 3*16*16
#define NBLK_  27

// Detect CDNA5 TDM path (device pass only).
#if defined(__HIP_DEVICE_COMPILE__)
#if __has_builtin(__builtin_amdgcn_tensor_load_to_lds) && __has_builtin(__builtin_amdgcn_s_wait_tensorcnt)
#define ATHENA_TDM 1
#endif
#endif

// ---------------------------------------------------------------------------
// Generic bf16 WMMA GEMM:  C[M,N](f32) = epi( A[M,Kp]bf16 @ B[N,Kp]bf16^T * alpha + bias )
// Grid: (Mpad/32, ceil(N/256), batch). Block: 256 threads = 8 waves (2 M x 4 N).
// Each wave computes a 16x64 strip = 4x v_wmma_f32_16x16x32_bf16 per K-step.
// A tiles (32x32 bf16 = 2 KB) staged in LDS via TENSOR_LOAD_TO_LDS (TDM),
// double-buffered so the DMA of tile s+1 overlaps the WMMAs of tile s.
// B streamed from global (weights are L2-resident: 192 MB L2 >> layer weights).
// Pad columns (n >= N) load a clamped row and are masked at the store, so the
// inner loop has no exec divergence.
// EPI: 0 = none, 1 = +bias, 2 = +bias then exact GELU.
// ---------------------------------------------------------------------------
template <int EPI>
__global__ __launch_bounds__(256) void k_gemm(
    const bf16* __restrict__ A, const bf16* __restrict__ Bm,
    const float* __restrict__ bias, float* __restrict__ C,
    int M, int N, int Kp, int ldc, float alpha,
    long long strideA, long long strideB, long long strideC) {
#if defined(__HIP_DEVICE_COMPILE__)
  __shared__ bf16 tA[2][32 * 32];

  const int tid  = threadIdx.x;
  const int lane = tid & 31;
  const int wid  = tid >> 5;
  const int waveM = wid >> 2;     // 0..1
  const int waveN = wid & 3;      // 0..3

  const bf16* Ab = A  + (long long)blockIdx.z * strideA;
  const bf16* Bb = Bm + (long long)blockIdx.z * strideB;
  float*      Cb = C  + (long long)blockIdx.z * strideC;

  const int m_wg = blockIdx.x * 32;
  const int n_wg = blockIdx.y * 256;
  const int Mrows = gridDim.x * 32;   // padded row count of A (tensor_dim1)
  const int nSteps = Kp >> 5;

  v8f acc[4];
#pragma unroll
  for (int t = 0; t < 4; ++t)
#pragma unroll
    for (int e = 0; e < 8; ++e) acc[t][e] = 0.f;

  // Per-lane fragment coordinates.
  const int arow = waveM * 16 + (lane & 15);   // A-frag row within LDS tile
  const int kk   = lane >> 4;                  // A-frag K-half selector
  const int nb0  = n_wg + waveN * 64 + (lane & 15);   // B-frag base column

#ifdef ATHENA_TDM
  // Issue a TDM load of A tile (32 rows x 32 K) for K-step `s` into LDS buffer `buf`.
  auto issue = [&](int s, int buf) {
    if (tid == 0) {
      unsigned long long ga =
          (unsigned long long)(uintptr_t)(Ab + (size_t)m_wg * Kp + (s << 5));
      const unsigned kpu = (unsigned)Kp, mru = (unsigned)Mrows;
      v4u g0;
      g0.x = 1u;                                        // count = 1 valid descriptor
      g0.y = (unsigned)(buf * 2048);                    // lds_addr (tA base = 0)
      g0.z = (unsigned)(ga & 0xffffffffull);            // global_addr[31:0]
      g0.w = (unsigned)((ga >> 32) & 0x1ffffffull)      // global_addr[56:32]
             | (2u << 30);                              // type = 2 ("image")
      v8i g1;
      g1[0] = (int)(1u << 16);                          // wg_mask=0, data_size=1 (2B)
      g1[1] = (int)((kpu & 0xffffu) << 16);             // tensor_dim0[15:0]
      g1[2] = (int)(((kpu >> 16) & 0xffffu)             // tensor_dim0[31:16]
                    | ((mru & 0xffffu) << 16));         // tensor_dim1[15:0]
      g1[3] = (int)(((mru >> 16) & 0xffffu)             // tensor_dim1[31:16]
                    | (32u << 16));                     // tile_dim0 = 32 (K)
      g1[4] = (int)32u;                                 // tile_dim1 = 32 rows, tile_dim2=0
      g1[5] = (int)kpu;                                 // tensor_dim0_stride[31:0]
      g1[6] = 0;
      g1[7] = 0;
      v4i gz4; gz4[0] = gz4[1] = gz4[2] = gz4[3] = 0;   // groups 2/3 unused (2D tensor)
      v8i gz8;
#pragma unroll
      for (int j = 0; j < 8; ++j) gz8[j] = 0;
      __builtin_amdgcn_tensor_load_to_lds(g0, g1, gz4, gz4, gz8, 0);
    }
  };

  issue(0, 0);
  __builtin_amdgcn_s_wait_tensorcnt(0);
  __syncthreads();

  for (int s = 0; s < nSteps; ++s) {
    const int cur = s & 1;
    if (s + 1 < nSteps) issue(s + 1, cur ^ 1);   // overlap DMA with compute

    // A fragment (16x32 bf16), lane half h holds K = 16h..16h+15 (contiguous) -> 2x ds_load_b128.
    const bf16* ar = &tA[cur][arow * 32];
    v16bf a;
#pragma unroll
    for (int j = 0; j < 8; ++j) {
      a[j]     = ar[kk * 16 + j];
      a[8 + j] = ar[kk * 16 + 8 + j];
    }

    const int k0 = s << 5;
#pragma unroll
    for (int t = 0; t < 4; ++t) {
      // B fragment (32x16): lane n, K = (lane/16)*16 + j -> contiguous 32 B.
      const int n  = nb0 + t * 16;
      const int nc = n < N ? n : (N - 1);        // clamp: pad cols masked at store
      const int kb = k0 + (lane >> 4) * 16;
      v16bf bv = *(const v16bf*)(Bb + (size_t)nc * Kp + kb);
      acc[t] = __builtin_amdgcn_wmma_f32_16x16x32_bf16(
          false, a, false, bv, (short)0, acc[t], false, false);
    }

    __builtin_amdgcn_s_wait_tensorcnt(0);        // tile s+1 landed
    __syncthreads();                             // all waves done with tile s
  }
#else
  for (int s = 0; s < nSteps; ++s) {
    const int k0 = s << 5;
    __syncthreads();
    {
      const int r = tid >> 3;            // 0..31 rows
      const int c = (tid & 7) * 4;       // 4 bf16 each
      const bf16* sp = Ab + (size_t)(m_wg + r) * Kp + (k0 + c);
#pragma unroll
      for (int j = 0; j < 4; ++j) tA[0][r * 32 + c + j] = sp[j];
    }
    __syncthreads();

    const bf16* ar = &tA[0][arow * 32];
    v16bf a;
#pragma unroll
    for (int j = 0; j < 8; ++j) {
      a[j]     = ar[kk * 16 + j];
      a[8 + j] = ar[kk * 16 + 8 + j];
    }
#pragma unroll
    for (int t = 0; t < 4; ++t) {
      const int n  = nb0 + t * 16;
      const int nc = n < N ? n : (N - 1);
      const int kb = k0 + (lane >> 4) * 16;
      v16bf bv = *(const v16bf*)(Bb + (size_t)nc * Kp + kb);
      acc[t] = __builtin_amdgcn_wmma_f32_16x16x32_bf16(
          false, a, false, bv, (short)0, acc[t], false, false);
    }
  }
#endif

  // Epilogue. C/D layout: VGPR r holds row M = r + 8*(lane/16); N = lane%16.
#pragma unroll
  for (int t = 0; t < 4; ++t) {
    const int n = nb0 + t * 16;
#pragma unroll
    for (int r = 0; r < 8; ++r) {
      const int m = m_wg + waveM * 16 + r + ((lane >> 4) << 3);
      if (m < M && n < N) {
        float v = acc[t][r] * alpha;
        if (EPI >= 1) v += bias[n];
        if (EPI == 2) v = 0.5f * v * (1.f + erff(v * 0.70710678118654752f));
        Cb[(size_t)m * ldc + n] = v;
      }
    }
  }
#endif  // __HIP_DEVICE_COMPILE__
}

// ---------------------------------------------------------------------------
// fp32 -> bf16 convert with zero padding. dst is [Mp x Kp]; src row stride ld.
// grid: (ceil(Mp*Kp/256), 1, batch)
// ---------------------------------------------------------------------------
__global__ void k_cvt(const float* __restrict__ src, bf16* __restrict__ dst,
                      int M, int K, int ld, int Kp, int Mp,
                      long long sbs, long long dbs) {
  long long idx = (long long)blockIdx.x * 256 + threadIdx.x;
  const long long tot = (long long)Mp * Kp;
  if (idx >= tot) return;
  const int r = (int)(idx / Kp), c = (int)(idx % Kp);
  const float* s = src + sbs * blockIdx.z;
  bf16* d = dst + dbs * blockIdx.z;
  const float v = (r < M && c < K) ? s[(size_t)r * ld + c] : 0.f;
  d[idx] = (bf16)v;
}

// dst[n,k] = src[k*ld + n] (transpose+convert+pad); dst is [Np x Kp]
__global__ void k_tcvt(const float* __restrict__ src, bf16* __restrict__ dst,
                       int R, int Ccols, int ld, int Np, int Kp,
                       long long sbs, long long dbs) {
  long long idx = (long long)blockIdx.x * 256 + threadIdx.x;
  const long long tot = (long long)Np * Kp;
  if (idx >= tot) return;
  const int n = (int)(idx / Kp), k = (int)(idx % Kp);
  const float* s = src + sbs * blockIdx.z;
  bf16* d = dst + dbs * blockIdx.z;
  const float v = (n < Ccols && k < R) ? s[(size_t)k * ld + n] : 0.f;
  d[idx] = (bf16)v;
}

// im2col for 16x16 stride-16 patches -> bf16 A [3136 x 768]
__global__ void k_im2col(const float* __restrict__ px, bf16* __restrict__ dst) {
  long long idx = (long long)blockIdx.x * 256 + threadIdx.x;
  const long long tot = (long long)MTOT * PK_;
  if (idx >= tot) return;
  const int f = (int)(idx % PK_);
  const int row = (int)(idx / PK_);
  const int b = row / SEQ, p = row % SEQ;
  const int ph = p / 28, pw = p % 28;
  const int c = f >> 8, rem = f & 255, dy = rem >> 4, dx = rem & 15;
  const size_t si = (((size_t)(b * 3 + c) * 448) + ph * 16 + dy) * 448 + pw * 16 + dx;
  dst[idx] = (bf16)px[si];
}

// patch_w [1152,3,2,16,16] temporal tap 0 -> bf16 [1152 x 768]
__global__ void k_cvt_patchw(const float* __restrict__ pw, bf16* __restrict__ dst) {
  long long idx = (long long)blockIdx.x * 256 + threadIdx.x;
  const long long tot = (long long)HID_ * PK_;
  if (idx >= tot) return;
  const int o = (int)(idx / PK_), f = (int)(idx % PK_);
  const int c = f >> 8, rem = f & 255;
  dst[idx] = (bf16)pw[(size_t)o * 1536 + (size_t)c * 512 + rem];
}

// x[b,p,h] += pos[p,h]
__global__ void k_addpos(float* __restrict__ x, const float* __restrict__ pos) {
  long long idx = (long long)blockIdx.x * 256 + threadIdx.x;
  const long long tot = (long long)MTOT * HID_;
  if (idx >= tot) return;
  const int h = (int)(idx % HID_);
  const int p = (int)((idx / HID_) % SEQ);
  x[idx] += pos[(size_t)p * HID_ + h];
}

__global__ void k_add(float* __restrict__ x, const float* __restrict__ y, long long n) {
  long long i = (long long)blockIdx.x * 256 + threadIdx.x;
  if (i < n) x[i] += y[i];
}

// Row LayerNorm (block per row), eps 1e-6
__global__ void k_ln(const float* __restrict__ x, const float* __restrict__ w,
                     const float* __restrict__ b, float* __restrict__ y, int L) {
  __shared__ float red[256];
  const int tid = threadIdx.x;
  const float* xr = x + (size_t)blockIdx.x * L;
  float* yr = y + (size_t)blockIdx.x * L;
  float s = 0.f;
  for (int j = tid; j < L; j += 256) s += xr[j];
  red[tid] = s; __syncthreads();
  for (int o = 128; o > 0; o >>= 1) { if (tid < o) red[tid] += red[tid + o]; __syncthreads(); }
  const float mu = red[0] / L;
  __syncthreads();
  float s2 = 0.f;
  for (int j = tid; j < L; j += 256) { const float d = xr[j] - mu; s2 += d * d; }
  red[tid] = s2; __syncthreads();
  for (int o = 128; o > 0; o >>= 1) { if (tid < o) red[tid] += red[tid + o]; __syncthreads(); }
  const float rstd = rsqrtf(red[0] / L + 1e-6f);
  for (int j = tid; j < L; j += 256) yr[j] = (xr[j] - mu) * rstd * w[j] + b[j];
}

// In-place row softmax (block per row)
__global__ void k_softmax(float* __restrict__ x, int L) {
  __shared__ float red[256];
  const int tid = threadIdx.x;
  float* xr = x + (size_t)blockIdx.x * L;
  float mx = -3.4e38f;
  for (int j = tid; j < L; j += 256) mx = fmaxf(mx, xr[j]);
  red[tid] = mx; __syncthreads();
  for (int o = 128; o > 0; o >>= 1) { if (tid < o) red[tid] = fmaxf(red[tid], red[tid + o]); __syncthreads(); }
  mx = red[0]; __syncthreads();
  float s = 0.f;
  for (int j = tid; j < L; j += 256) { const float e = __expf(xr[j] - mx); xr[j] = e; s += e; }
  red[tid] = s; __syncthreads();
  for (int o = 128; o > 0; o >>= 1) { if (tid < o) red[tid] += red[tid + o]; __syncthreads(); }
  const float inv = 1.f / red[0];
  for (int j = tid; j < L; j += 256) xr[j] *= inv;
}

// pixel shuffle: [B,784,1152] -> [B,196,4608]
__global__ void k_shuffle(const float* __restrict__ x, float* __restrict__ out) {
  long long idx = (long long)blockIdx.x * 256 + threadIdx.x;
  const long long tot = (long long)NBATCH * MERG * H4_;
  if (idx >= tot) return;
  const int h4 = (int)(idx % H4_);
  const int row = (int)(idx / H4_);
  const int b = row / MERG, m = row % MERG;
  const int mr = m / 14, mc = m % 14;
  const int q = h4 / HID_, h = h4 % HID_;
  const int r2 = q >> 1, c2 = q & 1;
  const int sidx = (mr * 2 + r2) * 28 + mc * 2 + c2;
  out[idx] = x[((size_t)(b * SEQ + sidx)) * HID_ + h];
}

// mean over 196 merged tokens -> out[b,4096]
__global__ void k_meanpool(const float* __restrict__ src, float* __restrict__ out) {
  const int o = blockIdx.x * 256 + threadIdx.x;
  const int b = blockIdx.y;
  if (o >= OUTD_) return;
  float s = 0.f;
  for (int i = 0; i < MERG; ++i) s += src[((size_t)(b * MERG + i)) * OUTD_ + o];
  out[(size_t)b * OUTD_ + o] = s * (1.f / MERG);
}

// ---------------------------------------------------------------------------
static inline unsigned cdiv(long long a, long long b) { return (unsigned)((a + b - 1) / b); }

extern "C" void kernel_launch(void* const* d_in, const int* in_sizes, int n_in,
                              void* d_out, int out_size, void* d_ws, size_t ws_size,
                              hipStream_t stream) {
  (void)in_sizes; (void)n_in; (void)out_size; (void)ws_size;
  // ---- inputs (setup_inputs dict order, depth-first) ----
  const float* pix     = (const float*)d_in[0];
  const float* patch_w = (const float*)d_in[1];
  const float* patch_b = (const float*)d_in[2];
  const float* pos     = (const float*)d_in[3];
  const float* n1w  = (const float*)d_in[4];
  const float* n1b  = (const float*)d_in[5];
  const float* qkvw = (const float*)d_in[6];
  const float* qkvb = (const float*)d_in[7];
  const float* pw   = (const float*)d_in[8];
  const float* pb   = (const float*)d_in[9];
  const float* n2w  = (const float*)d_in[10];
  const float* n2b  = (const float*)d_in[11];
  const float* f1w  = (const float*)d_in[12];
  const float* f1b  = (const float*)d_in[13];
  const float* f2w  = (const float*)d_in[14];
  const float* f2b  = (const float*)d_in[15];
  float* out = (float*)d_out;

  // ---- workspace bump allocator (256B aligned) ----
  size_t off = 0;
  auto alloc = [&](size_t bytes) -> void* {
    void* p = (char*)d_ws + off;
    off += (bytes + 255) & ~(size_t)255;
    return p;
  };
  bf16* pA   = (bf16*)alloc((size_t)MTOT * PK_ * 2);          // im2col bf16
  bf16* pWb  = (bf16*)alloc((size_t)HID_ * PK_ * 2);          // patch weight bf16
  float* x   = (float*)alloc((size_t)MTOT * HID_ * 4);        // residual stream
  float* y   = (float*)alloc((size_t)MTOT * HID_ * 4);        // LN output
  bf16* yb   = (bf16*)alloc((size_t)MTOT * HID_ * 2);         // LN output bf16
  bf16* wbuf = (bf16*)alloc((size_t)H4_ * H4_ * 2);           // shared weight-bf16 buffer
  float* qkv = (float*)alloc((size_t)MTOT * H3_ * 4);
  bf16* Qb   = (bf16*)alloc((size_t)NBATCH * SEQP * HID_ * 2);
  bf16* Kb   = (bf16*)alloc((size_t)NBATCH * SEQP * HID_ * 2);
  bf16* VT   = (bf16*)alloc((size_t)NBATCH * HID_ * SEQP * 2);
  float* sc  = (float*)alloc((size_t)NBATCH * SEQ * SEQ * 4);
  bf16* ab   = (bf16*)alloc((size_t)NBATCH * SEQP * SEQP * 2);
  float* ctx = (float*)alloc((size_t)MTOT * HID_ * 4);
  bf16* cb   = (bf16*)alloc((size_t)MTOT * HID_ * 2);
  float* hfc = (float*)alloc((size_t)MTOT * FI_ * 4);
  bf16* hb   = (bf16*)alloc((size_t)MTOT * FIP_ * 2);
  float* prj = (float*)alloc((size_t)MTOT * HID_ * 4);
  float* shf = (float*)alloc((size_t)MME * H4_ * 4);
  float* lnm = (float*)alloc((size_t)MME * H4_ * 4);
  bf16* mb   = (bf16*)alloc((size_t)MMEP * H4_ * 2);
  float* h1  = (float*)alloc((size_t)MME * H4_ * 4);
  bf16* h1b  = (bf16*)alloc((size_t)MMEP * H4_ * 2);
  float* mo  = (float*)alloc((size_t)MME * OUTD_ * 4);

  const float iscale = 1.0f / sqrtf((float)HID_);

#define G1(tot) dim3(cdiv((tot), 256))
#define G1B(tot, bz) dim3(cdiv((tot), 256), 1, (bz))

  // ================= patch embed =================
  k_im2col<<<G1((long long)MTOT * PK_), 256, 0, stream>>>(pix, pA);
  k_cvt_patchw<<<G1((long long)HID_ * PK_), 256, 0, stream>>>(patch_w, pWb);
  k_gemm<1><<<dim3(MTOT / 32, cdiv(HID_, 256), 1), 256, 0, stream>>>(
      pA, pWb, patch_b, x, MTOT, HID_, PK_, HID_, 1.f, 0, 0, 0);
  k_addpos<<<G1((long long)MTOT * HID_), 256, 0, stream>>>(x, pos);

  // merger runner
  auto run_merger = [&](const float* nw_, const float* nb_, const float* w1_,
                        const float* b1_, const float* w2_, const float* b2_,
                        bool postshuffle, float* outp) {
    const float* srcp;
    if (postshuffle) {
      k_shuffle<<<G1((long long)MME * H4_), 256, 0, stream>>>(x, shf);
      k_ln<<<MME, 256, 0, stream>>>(shf, nw_, nb_, lnm, H4_);
      srcp = lnm;
    } else {
      k_ln<<<MTOT, 256, 0, stream>>>(x, nw_, nb_, y, HID_);
      k_shuffle<<<G1((long long)MME * H4_), 256, 0, stream>>>(y, shf);
      srcp = shf;
    }
    k_cvt<<<G1((long long)MMEP * H4_), 256, 0, stream>>>(srcp, mb, MME, H4_, H4_, H4_, MMEP, 0, 0);
    k_cvt<<<G1((long long)H4_ * H4_), 256, 0, stream>>>(w1_, wbuf, H4_, H4_, H4_, H4_, H4_, 0, 0);
    k_gemm<2><<<dim3(MMEP / 32, cdiv(H4_, 256), 1), 256, 0, stream>>>(
        mb, wbuf, b1_, h1, MME, H4_, H4_, H4_, 1.f, 0, 0, 0);
    k_cvt<<<G1((long long)MMEP * H4_), 256, 0, stream>>>(h1, h1b, MME, H4_, H4_, H4_, MMEP, 0, 0);
    k_cvt<<<G1((long long)OUTD_ * H4_), 256, 0, stream>>>(w2_, wbuf, OUTD_, H4_, H4_, H4_, OUTD_, 0, 0);
    k_gemm<1><<<dim3(MMEP / 32, cdiv(OUTD_, 256), 1), 256, 0, stream>>>(
        h1b, wbuf, b2_, mo, MME, OUTD_, H4_, OUTD_, 1.f, 0, 0, 0);
    k_meanpool<<<dim3(cdiv(OUTD_, 256), NBATCH), 256, 0, stream>>>(mo, outp);
  };

  // ================= 27 transformer blocks =================
  for (int i = 0; i < NBLK_; ++i) {
    // ---- attention ----
    k_ln<<<MTOT, 256, 0, stream>>>(x, n1w + (size_t)i * HID_, n1b + (size_t)i * HID_, y, HID_);
    k_cvt<<<G1((long long)MTOT * HID_), 256, 0, stream>>>(y, yb, MTOT, HID_, HID_, HID_, MTOT, 0, 0);
    k_cvt<<<G1((long long)H3_ * HID_), 256, 0, stream>>>(
        qkvw + (size_t)i * H3_ * HID_, wbuf, H3_, HID_, HID_, HID_, H3_, 0, 0);
    k_gemm<1><<<dim3(MTOT / 32, cdiv(H3_, 256), 1), 256, 0, stream>>>(
        yb, wbuf, qkvb + (size_t)i * H3_, qkv, MTOT, H3_, HID_, H3_, 1.f, 0, 0, 0);

    // split -> padded bf16 Q, K and transposed V (per image, grid.z = 4)
    k_cvt<<<G1B((long long)SEQP * HID_, NBATCH), 256, 0, stream>>>(
        qkv + 0, Qb, SEQ, HID_, H3_, HID_, SEQP,
        (long long)SEQ * H3_, (long long)SEQP * HID_);
    k_cvt<<<G1B((long long)SEQP * HID_, NBATCH), 256, 0, stream>>>(
        qkv + HID_, Kb, SEQ, HID_, H3_, HID_, SEQP,
        (long long)SEQ * H3_, (long long)SEQP * HID_);
    k_tcvt<<<G1B((long long)HID_ * SEQP, NBATCH), 256, 0, stream>>>(
        qkv + 2 * HID_, VT, SEQ, HID_, H3_, HID_, SEQP,
        (long long)SEQ * H3_, (long long)HID_ * SEQP);

    // scores = softmax(Q K^T / sqrt(H))
    k_gemm<0><<<dim3(SEQP / 32, cdiv(SEQ, 256), NBATCH), 256, 0, stream>>>(
        Qb, Kb, nullptr, sc, SEQ, SEQ, HID_, SEQ, iscale,
        (long long)SEQP * HID_, (long long)SEQP * HID_, (long long)SEQ * SEQ);
    k_softmax<<<NBATCH * SEQ, 256, 0, stream>>>(sc, SEQ);
    k_cvt<<<G1B((long long)SEQP * SEQP, NBATCH), 256, 0, stream>>>(
        sc, ab, SEQ, SEQ, SEQ, SEQP, SEQP,
        (long long)SEQ * SEQ, (long long)SEQP * SEQP);

    // ctx = attn @ V
    k_gemm<0><<<dim3(SEQP / 32, cdiv(HID_, 256), NBATCH), 256, 0, stream>>>(
        ab, VT, nullptr, ctx, SEQ, HID_, SEQP, HID_, 1.f,
        (long long)SEQP * SEQP, (long long)HID_ * SEQP, (long long)SEQ * HID_);

    // proj + residual
    k_cvt<<<G1((long long)MTOT * HID_), 256, 0, stream>>>(ctx, cb, MTOT, HID_, HID_, HID_, MTOT, 0, 0);
    k_cvt<<<G1((long long)HID_ * HID_), 256, 0, stream>>>(
        pw + (size_t)i * HID_ * HID_, wbuf, HID_, HID_, HID_, HID_, HID_, 0, 0);
    k_gemm<1><<<dim3(MTOT / 32, cdiv(HID_, 256), 1), 256, 0, stream>>>(
        cb, wbuf, pb + (size_t)i * HID_, prj, MTOT, HID_, HID_, HID_, 1.f, 0, 0, 0);
    k_add<<<G1((long long)MTOT * HID_), 256, 0, stream>>>(x, prj, (long long)MTOT * HID_);

    // ---- MLP ----
    k_ln<<<MTOT, 256, 0, stream>>>(x, n2w + (size_t)i * HID_, n2b + (size_t)i * HID_, y, HID_);
    k_cvt<<<G1((long long)MTOT * HID_), 256, 0, stream>>>(y, yb, MTOT, HID_, HID_, HID_, MTOT, 0, 0);
    k_cvt<<<G1((long long)FI_ * HID_), 256, 0, stream>>>(
        f1w + (size_t)i * FI_ * HID_, wbuf, FI_, HID_, HID_, HID_, FI_, 0, 0);
    k_gemm<2><<<dim3(MTOT / 32, cdiv(FI_, 256), 1), 256, 0, stream>>>(
        yb, wbuf, f1b + (size_t)i * FI_, hfc, MTOT, FI_, HID_, FI_, 1.f, 0, 0, 0);
    k_cvt<<<G1((long long)MTOT * FIP_), 256, 0, stream>>>(hfc, hb, MTOT, FI_, FI_, FIP_, MTOT, 0, 0);
    k_cvt<<<G1((long long)HID_ * FIP_), 256, 0, stream>>>(
        f2w + (size_t)i * HID_ * FI_, wbuf, HID_, FI_, FI_, FIP_, HID_, 0, 0);
    k_gemm<1><<<dim3(MTOT / 32, cdiv(HID_, 256), 1), 256, 0, stream>>>(
        hb, wbuf, f2b + (size_t)i * HID_, prj, MTOT, HID_, FIP_, HID_, 1.f, 0, 0, 0);
    k_add<<<G1((long long)MTOT * HID_), 256, 0, stream>>>(x, prj, (long long)MTOT * HID_);

    // ---- deepstack mergers on hs[8], hs[16], hs[24] (post-block states) ----
    if (i == 8 || i == 16 || i == 24) {
      const int j = (i == 8) ? 0 : (i == 16) ? 1 : 2;
      const int base = 22 + 6 * j;
      run_merger((const float*)d_in[base + 0], (const float*)d_in[base + 1],
                 (const float*)d_in[base + 2], (const float*)d_in[base + 3],
                 (const float*)d_in[base + 4], (const float*)d_in[base + 5],
                 /*postshuffle=*/true, out + (size_t)(1 + j) * NBATCH * OUTD_);
    }
  }

  // ================= final merger (pre-shuffle LN) =================
  run_merger((const float*)d_in[16], (const float*)d_in[17], (const float*)d_in[18],
             (const float*)d_in[19], (const float*)d_in[20], (const float*)d_in[21],
             /*postshuffle=*/false, out);
#undef G1
#undef G1B
}
